// InCritic_22531398434854
// MI455X (gfx1250) — matrix operand order repaired and checked by reading, not compiled, not run on hardware
//
#include <hip/hip_runtime.h>

typedef _Float16 v16h __attribute__((ext_vector_type(16)));
typedef float    v8f  __attribute__((ext_vector_type(8)));
typedef unsigned int u32x4 __attribute__((ext_vector_type(4)));
typedef int          i32x8 __attribute__((ext_vector_type(8)));
typedef int          i32x4 __attribute__((ext_vector_type(4)));

// ---------------------------------------------------------------------------
// Weight fragment store (in d_ws): each 32x16 (KxN) tile of a weight matrix is
// stored as 512 f16: [lane 0..31][16 halfs].  Per the CDNA5 B-matrix layout
// (16-bit, 32x16): lane L holds column N = L%16; lanes 0-15 hold K=0..7,16..23,
// lanes 16-31 hold K=8..15,24..31, ascending K within the 16 halfs.
// Sets that are consumed together are contiguous so one TDM DMA fills each.
// ---------------------------------------------------------------------------
enum { FR_MPW1 = 0, FR_MPW2 = 16, FR_EAWV = 32, FR_PHIW1 = 34, FR_PHIW2 = 66,
       FR_PHIW3 = 98, FR_PHIW4 = 130, FR_NAWV = 162, FR_RHOW1 = 170,
       FR_RHOW3 = 202, N_FRAGS = 234 };

// cache-relative bases
enum { CB_MPW1 = 0, CB_MPW2 = 16, CB_EAWV = 32, CB_PHI1 = 0, CB_PHI2 = 32,
       CB_NA = 0, CB_RHO = 8 };

struct FragDesc { int base, K, N, KT, NT; };
__constant__ FragDesc kFrags[10] = {
  {FR_MPW1,    8, 256, 1, 16},
  {FR_MPW2,  256,  32, 8,  2},
  {FR_EAWV,   32,  32, 1,  2},
  {FR_PHIW1,  61, 256, 2, 16},
  {FR_PHIW2, 256,  64, 8,  4},
  {FR_PHIW3,  61, 256, 2, 16},
  {FR_PHIW4, 256,  64, 8,  4},
  {FR_NAWV,   64,  64, 2,  4},
  {FR_RHOW1,  64, 256, 2, 16},
  {FR_RHOW3,  64, 256, 2, 16},
};

struct WPtrs { const float* p[10]; };

__global__ void prep_frags(WPtrs w, _Float16* frags) {
  int f = blockIdx.x;
  int d = 0;
  #pragma unroll
  for (int i = 0; i < 10; ++i)
    if (f >= kFrags[i].base) d = i;
  FragDesc fd = kFrags[d];
  int local = f - fd.base;
  int kt = local / fd.NT, nt = local % fd.NT;
  int lane = threadIdx.x & 31;
  int n  = nt * 16 + (lane & 15);
  int kb = kt * 32 + ((lane >> 4) << 3);
  const float* W = w.p[d];
  union { _Float16 h[16]; v16h v; } u;
  #pragma unroll
  for (int i = 0; i < 16; ++i) {
    int k = kb + (i < 8 ? i : i + 8);
    u.h[i] = (k < fd.K) ? (_Float16)W[k * fd.N + n] : (_Float16)0.f;
  }
  *(v16h*)(frags + (size_t)f * 512 + lane * 16) = u.v;
}

// ---------------------------------------------------------------------------
// Graph constants
// ---------------------------------------------------------------------------
__constant__ int cEDG[6][2]  = {{0,1},{0,2},{1,0},{1,2},{2,0},{2,1}};
__constant__ int cPRED[6][2] = {{0,3},{1,5},{0,4},{2,7},{1,6},{2,8}};
__constant__ int cINC[3][2]  = {{2,4},{0,5},{1,3}};

struct Params {
  const float *obs, *act, *ag, *g;
  const _Float16* frags;
  const float *mp_b1, *mp_b2;
  const float *ea_Wq, *ea_bq, *ea_Wk, *ea_bk, *ea_bv;
  const float *phi_b1, *phi_b2, *phi_b3, *phi_b4;
  const float *na_Wq, *na_bq, *na_Wk, *na_bk, *na_bv;
  const float *rho_b1, *rho_W2, *rho_b2, *rho_b3, *rho_W4, *rho_b4;
  float* out;
  int B;
};

// ---------------------------------------------------------------------------
// TDM fill: DMA `nfrags` KB of pre-swizzled fragments global -> LDS cache.
// 1-D descriptor: data_size=4B, tile_dim0 = tensor_dim0 = nfrags*256 dwords.
// Issued by wave 0 only (TDM ignores EXEC; one op per instruction).
// This toolchain exposes the 6-arg builtin: (g0, g1, g2, g3, g1ext, cpol).
// ---------------------------------------------------------------------------
__device__ inline void tdm_fill(_Float16* lds_dst, const _Float16* gsrc,
                                int nfrags) {
#if __has_builtin(__builtin_amdgcn_tensor_load_to_lds)
  if (threadIdx.x < 32) {
    unsigned lds_addr = (unsigned)(size_t)lds_dst;        // LDS byte offset
    unsigned long long ga = (unsigned long long)(size_t)gsrc;
    unsigned nd = (unsigned)nfrags * 256u;                // dwords
    u32x4 g0;
    g0[0] = 1u;                                           // count=1, user D#
    g0[1] = lds_addr;                                     // lds_addr[31:0]
    g0[2] = (unsigned)ga;                                 // global_addr[31:0]
    g0[3] = ((unsigned)(ga >> 32) & 0x01FFFFFFu) | (2u << 30); // [56:32]|type=2
    i32x8 g1;
    g1[0] = (int)(2u << 16);                              // data_size = 4B
    g1[1] = (int)((nd & 0xFFFFu) << 16);                  // tensor_dim0[15:0]
    g1[2] = (int)((nd >> 16) | (1u << 16));               // td0[31:16]|td1=1
    g1[3] = (int)(nd << 16);                              // tile_dim0 = nd
    g1[4] = 1;                                            // tile_dim1=1
    g1[5] = (int)nd;                                      // td0_stride[31:0]
    g1[6] = 0;
    g1[7] = 0;
    i32x4 z4 = {0, 0, 0, 0};
    i32x8 z8 = {0, 0, 0, 0, 0, 0, 0, 0};
    __builtin_amdgcn_tensor_load_to_lds(g0, g1, z4, z4, z8, 0);
    __builtin_amdgcn_s_wait_tensorcnt(0);
  }
#else
  {
    const uint4* s = (const uint4*)gsrc;
    uint4* d = (uint4*)lds_dst;
    for (int i = threadIdx.x; i < nfrags * 64; i += 64) d[i] = s[i];
  }
#endif
  __syncthreads();
}

// A-fragment (16x32 f16, M x K) from row-major LDS staging [16][ld] halfs.
__device__ inline v16h load_a(const _Float16* s, int ld, int kt, int lane) {
  int m  = lane & 15;
  int kh = ((lane >> 4) << 3) + kt * 32;
  union { v16h v; unsigned u[8]; } r;
  const unsigned* row = (const unsigned*)(s + m * ld);
  #pragma unroll
  for (int i = 0; i < 8; ++i) {
    int k = ((i < 4) ? 2 * i : 2 * i + 8) + kh;
    r.u[i] = row[k >> 1];
  }
  return r.v;
}

__device__ inline v16h load_b(const _Float16* bc, int idx, int lane) {
  return *(const v16h*)(bc + (size_t)idx * 512 + lane * 16);
}

template<int KT, int NT>
__device__ inline void gemm(const _Float16* A, int ldA,
                            const _Float16* bc, int fbase,
                            v8f acc[NT], int lane) {
  #pragma unroll
  for (int kt = 0; kt < KT; ++kt) {
    v16h a = load_a(A, ldA, kt, lane);
    #pragma unroll
    for (int nt = 0; nt < NT; ++nt) {
      v16h b = load_b(bc, fbase + kt * NT + nt, lane);
      acc[nt] = __builtin_amdgcn_wmma_f32_16x16x32_f16(
          false, a, false, b, (short)0, acc[nt], false, false);
    }
  }
}

// C-fragment epilogue: lane L holds column N=L%16 (+16*nt), rows M=v+8*(L/16).
__device__ inline void store_tile(_Float16* dst, int ld, int nt, v8f acc,
                                  const float* bias, float bscale, bool relu,
                                  int lane) {
  int col = nt * 16 + (lane & 15);
  float bb = bias[col] * bscale;
  int mb = (lane >> 4) << 3;
  #pragma unroll
  for (int v = 0; v < 8; ++v) {
    float x = acc[v] + bb;
    if (relu) x = fmaxf(x, 0.f);
    dst[(mb + v) * ld + col] = (_Float16)x;
  }
}

// ---------------------------------------------------------------------------
// Fused critic: 2 waves/block, each wave handles 16 batch rows.
// Per-wave LDS (halfs): S0[4096]=16x256 staging (h / xbar / hbar / s / scratch)
//                       S1[3072]=48x64 phi-input, S2[3072]=48x64 phi output,
//                       S3[3072]=16x6x32 edge-MLP output.
// Block-shared: fcache[32768] = 64KB weight-fragment cache (TDM-filled).
// ---------------------------------------------------------------------------
__global__ __launch_bounds__(64) void critic_kernel(Params p) {
  const int lane = threadIdx.x & 31;
  const int wave = threadIdx.x >> 5;
  const int batch0 = (blockIdx.x * 2 + wave) * 16;

  __shared__ _Float16 smem[2][13312];
  __shared__ _Float16 fcache[32768];
  _Float16* S0 = smem[wave];
  _Float16* S1 = S0 + 4096;
  _Float16* S2 = S1 + 3072;
  _Float16* S3 = S2 + 3072;

  auto gb = [&](int b) { int v = batch0 + b; return v < p.B ? v : p.B - 1; };

  // ========================= edge MLP (mp) ==================================
  tdm_fill(fcache, p.frags + (size_t)FR_MPW1 * 512, 34);   // mpW1|mpW2|eaWv

  for (int e = 0; e < 6; ++e) {
    if (lane < 16) {                           // stage 16x32 input rows in S1
      int b = gb(lane);
      const float* ob = p.obs + (size_t)b * 55;
      int p0 = cPRED[e][0], p1 = cPRED[e][1];
      int o0 = cEDG[e][0],  o1 = cEDG[e][1];
      _Float16* r = S1 + lane * 32;
      r[0] = (_Float16)(p.g[b * 9 + p0] - p.ag[b * 9 + p0]);
      r[1] = (_Float16)(p.g[b * 9 + p1] - p.ag[b * 9 + p1]);
      #pragma unroll
      for (int j = 0; j < 3; ++j) {
        r[2 + j] = (_Float16)ob[10 + o0 * 15 + j];
        r[5 + j] = (_Float16)ob[10 + o1 * 15 + j];
      }
      for (int k = 8; k < 32; ++k) r[k] = (_Float16)0.f;
    }
    __syncthreads();
    v8f acc[16] = {};
    gemm<1, 16>(S1, 32, fcache, CB_MPW1, acc, lane);
    __syncthreads();
    #pragma unroll
    for (int nt = 0; nt < 16; ++nt)
      store_tile(S0, 256, nt, acc[nt], p.mp_b1, 1.f, true, lane);   // h=relu
    __syncthreads();
    v8f a2[2] = {};
    gemm<8, 2>(S0, 256, fcache, CB_MPW2, a2, lane);
    #pragma unroll
    for (int nt = 0; nt < 2; ++nt)                                  // out_mp
      store_tile(S3 + e * 32, 192, nt, a2[nt], p.mp_b2, 1.f, false, lane);
    __syncthreads();
  }

  // ===================== edge attention (q/k dim==1, scale==1) ==============
  // sum over tokens collapses: efeat = (w0*x0 + w1*x1) @ Wv + 2*bv
  for (int pi = lane; pi < 48; pi += 32) {
    int n = pi >> 4, b = pi & 15;
    const _Float16* x0 = S3 + b * 192 + cINC[n][0] * 32;
    const _Float16* x1 = S3 + b * 192 + cINC[n][1] * 32;
    float q0 = p.ea_bq[0], q1 = p.ea_bq[0];
    float k0 = p.ea_bk[0], k1 = p.ea_bk[0];
    for (int j = 0; j < 32; ++j) {
      float a0 = (float)x0[j], a1 = (float)x1[j];
      float wq = p.ea_Wq[j], wk = p.ea_Wk[j];
      q0 += a0 * wq; q1 += a1 * wq;
      k0 += a0 * wk; k1 += a1 * wk;
    }
    float s00 = q0 * k0, s01 = q0 * k1, s10 = q1 * k0, s11 = q1 * k1;
    float m0 = fmaxf(s00, s01), m1 = fmaxf(s10, s11);
    float e00 = __expf(s00 - m0), e01 = __expf(s01 - m0);
    float e10 = __expf(s10 - m1), e11 = __expf(s11 - m1);
    float i0 = 1.f / (e00 + e01), i1 = 1.f / (e10 + e11);
    float w0 = e00 * i0 + e10 * i1, w1 = e01 * i0 + e11 * i1;
    _Float16* xb = S0 + pi * 32;                 // xbar rows (node-major)
    for (int j = 0; j < 32; ++j)
      xb[j] = (_Float16)(w0 * (float)x0[j] + w1 * (float)x1[j]);
  }

  // ================== phi-input assembly (cols 0..28, pad 61..63) ===========
  for (int r = lane; r < 48; r += 32) {
    int n = r >> 4, b = gb(r & 15);
    _Float16* row = S1 + r * 64;
    const float* A = p.act + (size_t)b * 4;
    const float* O = p.obs + (size_t)b * 55;
    #pragma unroll
    for (int j = 0; j < 4; ++j)  row[j]      = (_Float16)A[j];
    #pragma unroll
    for (int j = 0; j < 10; ++j) row[4 + j]  = (_Float16)O[j];
    #pragma unroll
    for (int j = 0; j < 15; ++j) row[14 + j] = (_Float16)O[10 + n * 15 + j];
    row[61] = row[62] = row[63] = (_Float16)0.f;
  }
  __syncthreads();

  // edge_feat GEMM -> phi-input cols 29..60
  for (int mt = 0; mt < 3; ++mt) {
    v8f acc[2] = {};
    gemm<1, 2>(S0 + mt * 16 * 32, 32, fcache, CB_EAWV, acc, lane);
    #pragma unroll
    for (int nt = 0; nt < 2; ++nt)
      store_tile(S1 + mt * 16 * 64 + 29, 64, nt, acc[nt], p.ea_bv, 2.f, false, lane);
  }
  __syncthreads();

  // ========================= two critic branches ============================
  for (int br = 0; br < 2; ++br) {
    const int fW1 = br ? FR_PHIW3 : FR_PHIW1;    // phiWx2 is contiguous after
    const int fR1 = br ? FR_RHOW3 : FR_RHOW1;
    const float* pb1 = br ? p.phi_b3 : p.phi_b1;
    const float* pb2 = br ? p.phi_b4 : p.phi_b2;
    const float* rb1 = br ? p.rho_b3 : p.rho_b1;
    const float* rW2 = br ? p.rho_W4 : p.rho_W2;
    const float* rb2 = br ? p.rho_b4 : p.rho_b2;

    tdm_fill(fcache, p.frags + (size_t)fW1 * 512, 64);     // phiW{1,2} pair

    // phi: 48x61 -> relu 256 -> relu 64, per 16-row M-tile
    for (int mt = 0; mt < 3; ++mt) {
      v8f acc[16] = {};
      gemm<2, 16>(S1 + mt * 16 * 64, 64, fcache, CB_PHI1, acc, lane);
      __syncthreads();
      #pragma unroll
      for (int nt = 0; nt < 16; ++nt)
        store_tile(S0, 256, nt, acc[nt], pb1, 1.f, true, lane);
      __syncthreads();
      v8f a2[4] = {};
      gemm<8, 4>(S0, 256, fcache, CB_PHI2, a2, lane);
      #pragma unroll
      for (int nt = 0; nt < 4; ++nt)
        store_tile(S2 + mt * 16 * 64, 64, nt, a2[nt], pb2, 1.f, true, lane);
      __syncthreads();
    }

    tdm_fill(fcache, p.frags + (size_t)FR_NAWV * 512, 8);  // naWv
    tdm_fill(fcache + (size_t)8 * 512, p.frags + (size_t)fR1 * 512, 32);

    // node attention (q/k dim==1, scale==1) -> hbar = sum_t w_t h_t
    if (lane < 16) {
      int b = lane;
      float q[3], k[3];
      #pragma unroll
      for (int t = 0; t < 3; ++t) {
        const _Float16* h = S2 + (t * 16 + b) * 64;
        float qq = p.na_bq[0], kk = p.na_bk[0];
        for (int j = 0; j < 64; ++j) {
          float x = (float)h[j];
          qq += x * p.na_Wq[j];
          kk += x * p.na_Wk[j];
        }
        q[t] = qq; k[t] = kk;
      }
      float w[3] = {0.f, 0.f, 0.f};
      #pragma unroll
      for (int s = 0; s < 3; ++s) {
        float t0 = q[s] * k[0], t1 = q[s] * k[1], t2 = q[s] * k[2];
        float m = fmaxf(t0, fmaxf(t1, t2));
        float e0 = __expf(t0 - m), e1 = __expf(t1 - m), e2 = __expf(t2 - m);
        float inv = 1.f / (e0 + e1 + e2);
        w[0] += e0 * inv; w[1] += e1 * inv; w[2] += e2 * inv;
      }
      _Float16* hb = S0 + b * 64;
      for (int j = 0; j < 64; ++j) {
        float s = w[0] * (float)S2[(0 * 16 + b) * 64 + j]
                + w[1] * (float)S2[(1 * 16 + b) * 64 + j]
                + w[2] * (float)S2[(2 * 16 + b) * 64 + j];
        hb[j] = (_Float16)s;
      }
    }
    __syncthreads();

    // s = hbar @ na_Wv + 3*na_bv   (16x64 @ 64x64)
    {
      v8f a2[4] = {};
      gemm<2, 4>(S0, 64, fcache, CB_NA, a2, lane);
      #pragma unroll
      for (int nt = 0; nt < 4; ++nt)
        store_tile(S0 + 2048, 64, nt, a2[nt], p.na_bv, 3.f, false, lane);
    }
    __syncthreads();

    // rho: relu(s@W1+b1) with the 256->1 dot fused into the epilogue
    {
      v8f acc[16] = {};
      gemm<2, 16>(S0 + 2048, 64, fcache, CB_RHO, acc, lane);
      float part[8] = {0.f, 0.f, 0.f, 0.f, 0.f, 0.f, 0.f, 0.f};
      #pragma unroll
      for (int nt = 0; nt < 16; ++nt) {
        int col = nt * 16 + (lane & 15);
        float b1v = rb1[col], w2 = rW2[col];
        #pragma unroll
        for (int v = 0; v < 8; ++v)
          part[v] += fmaxf(acc[nt][v] + b1v, 0.f) * w2;
      }
      __syncthreads();
      float* Sf = (float*)S0;                    // hbar region is dead now
      #pragma unroll
      for (int v = 0; v < 8; ++v) Sf[lane * 8 + v] = part[v];
      __syncthreads();
      if (lane < 16) {
        int b = lane;
        int lb = (b >= 8) ? 16 : 0;
        int v = b & 7;
        float s = rb2[0];
        for (int l = 0; l < 16; ++l) s += Sf[(lb + l) * 8 + v];
        int g = batch0 + b;
        if (g < p.B) p.out[(size_t)br * p.B + g] = s;
      }
      __syncthreads();
    }
  }
}

// ---------------------------------------------------------------------------
extern "C" void kernel_launch(void* const* d_in, const int* in_sizes, int n_in,
                              void* d_out, int out_size, void* d_ws, size_t ws_size,
                              hipStream_t stream) {
  (void)n_in; (void)out_size; (void)ws_size;
  const int B = in_sizes[0] / 55;
  _Float16* frags = (_Float16*)d_ws;

  WPtrs w;
  w.p[0] = (const float*)d_in[4];   // mp_W1
  w.p[1] = (const float*)d_in[6];   // mp_W2
  w.p[2] = (const float*)d_in[12];  // ea_Wv
  w.p[3] = (const float*)d_in[14];  // phi_W1
  w.p[4] = (const float*)d_in[16];  // phi_W2
  w.p[5] = (const float*)d_in[18];  // phi_W3
  w.p[6] = (const float*)d_in[20];  // phi_W4
  w.p[7] = (const float*)d_in[26];  // na_Wv
  w.p[8] = (const float*)d_in[28];  // rho_W1
  w.p[9] = (const float*)d_in[32];  // rho_W3
  prep_frags<<<N_FRAGS, 32, 0, stream>>>(w, frags);

  Params p;
  p.obs    = (const float*)d_in[0];
  p.act    = (const float*)d_in[1];
  p.ag     = (const float*)d_in[2];
  p.g      = (const float*)d_in[3];
  p.frags  = frags;
  p.mp_b1  = (const float*)d_in[5];
  p.mp_b2  = (const float*)d_in[7];
  p.ea_Wq  = (const float*)d_in[8];
  p.ea_bq  = (const float*)d_in[9];
  p.ea_Wk  = (const float*)d_in[10];
  p.ea_bk  = (const float*)d_in[11];
  p.ea_bv  = (const float*)d_in[13];
  p.phi_b1 = (const float*)d_in[15];
  p.phi_b2 = (const float*)d_in[17];
  p.phi_b3 = (const float*)d_in[19];
  p.phi_b4 = (const float*)d_in[21];
  p.na_Wq  = (const float*)d_in[22];
  p.na_bq  = (const float*)d_in[23];
  p.na_Wk  = (const float*)d_in[24];
  p.na_bk  = (const float*)d_in[25];
  p.na_bv  = (const float*)d_in[27];
  p.rho_b1 = (const float*)d_in[29];
  p.rho_W2 = (const float*)d_in[30];
  p.rho_b2 = (const float*)d_in[31];
  p.rho_b3 = (const float*)d_in[33];
  p.rho_W4 = (const float*)d_in[34];
  p.rho_b4 = (const float*)d_in[35];
  p.out    = (float*)d_out;
  p.B      = B;

  const int blocks = (B + 31) / 32;
  critic_kernel<<<blocks, 64, 0, stream>>>(p);
}
